// PositionAttention_63050119905889
// MI455X (gfx1250) — matrix operand order repaired and checked by reading, not compiled, not run on hardware
//
#include <hip/hip_runtime.h>
#include <hip/hip_bf16.h>

// ---------------------------------------------------------------------------
// Problem constants (from reference): B=4, C=256, Ci=32, H=W=64, N=4096
// ---------------------------------------------------------------------------
#define BB    4
#define CC    256
#define CI    32
#define NN    4096
#define NT16  256   // N / 16
#define NT32  128   // N / 32

typedef __attribute__((ext_vector_type(16))) __bf16 v16bf;
typedef __attribute__((ext_vector_type(8)))  float  v8f;

// Workspace layout (bf16 element offsets). Each "chunk" is one 512-element
// WMMA operand tile (16x32 A-layout or 32x16 B-layout), lane l owns elements
// [l*16, l*16+16) -> one 32-byte contiguous load per lane.
//   XB : x as B-operand tiles        [B][8 ksteps][NT16]  (32c  x 16n)
//   WC : Wq/Wk/Wv as A-operand tiles [160 chunks]         (16o  x 32c)
//   QB : Q as B-operand tiles        [B][NT16]            (32ci x 16q)
//   KA : K as A-operand tiles        [B][NT16]            (16key x 32ci)
//   VB : V^T as B-operand tiles      [B][NT32][16 ctile]  (32key x 16c)
#define XB_OFF  ((size_t)0)
#define XB_ELM  ((size_t)BB * 8 * NT16 * 512)            // 4,194,304
#define WC_OFF  (XB_OFF + XB_ELM)
#define WC_ELM  ((size_t)160 * 512)                      // 81,920
#define QB_OFF  (WC_OFF + WC_ELM)
#define QB_ELM  ((size_t)BB * NT16 * 512)                // 524,288
#define KA_OFF  (QB_OFF + QB_ELM)
#define KA_ELM  QB_ELM
#define VB_OFF  (KA_OFF + KA_ELM)
#define VB_ELM  ((size_t)BB * NT32 * 16 * 512)           // 4,194,304

// A-operand (16x32, 16-bit) slot map per ISA 7.12.2: lane = arow + (hh?16:0),
// holds K in [hh,hh+8) U [16+hh,16+hh+8), hh = 0 or 8.
__device__ __forceinline__ int a_slot(int arow, int K) {
    int kl = K & 15;
    int hh = (kl >= 8) ? 8 : 0;
    int lanep = arow + (hh ? 16 : 0);
    int s = (K < 16) ? (K - hh) : (8 + (K - 16 - hh));
    return lanep * 16 + s;
}
// B-operand (32x16, 16-bit): lanes 0-15 hold K=0..7 U 16..23 (col=lane), lanes 16-31
// hold K=8..15 U 24..31.
__device__ __forceinline__ int b_slot(int K, int col) {
    int lanep = col + ((K & 8) ? 16 : 0);
    return lanep * 16 + (K & 7) + ((K & 16) ? 8 : 0);
}

// ---------------------------------------------------------------------------
// Kernel 0a: convert 1x1-conv weights (f32 [o,c]) into bf16 A-operand chunks.
// Chunk order: Wq mt0..1 x ks0..7 (16), Wk (16), Wv mt0..15 x ks0..7 (128).
// ---------------------------------------------------------------------------
__global__ void conv_w_kernel(const float* __restrict__ Wq,
                              const float* __restrict__ Wk,
                              const float* __restrict__ Wv,
                              __bf16* __restrict__ ws) {
    int gid = blockIdx.x * blockDim.x + threadIdx.x;
    if (gid >= (int)WC_ELM) return;
    int chunk = gid >> 9;
    int flat  = gid & 511;
    int lanep = flat >> 4, s = flat & 15;
    int hh  = (lanep < 16) ? 0 : 8;
    int K   = (s < 8) ? (hh + s) : (16 + hh + (s - 8));
    int row = lanep & 15;
    const float* W; int mt, ks;
    if (chunk < 16)      { W = Wq; mt = chunk >> 3;        ks = chunk & 7; }
    else if (chunk < 32) { W = Wk; mt = (chunk - 16) >> 3; ks = (chunk - 16) & 7; }
    else                 { W = Wv; mt = (chunk - 32) >> 3; ks = (chunk - 32) & 7; }
    ws[WC_OFF + gid] = (__bf16)W[(mt * 16 + row) * CC + ks * 32 + K];
}

// ---------------------------------------------------------------------------
// Kernel 0b: convert x (f32 [B,C,N]) into bf16 B-operand chunks (32c x 16n).
// ---------------------------------------------------------------------------
__global__ void conv_x_kernel(const float* __restrict__ x,
                              __bf16* __restrict__ ws) {
    int gid = blockIdx.x * blockDim.x + threadIdx.x;
    if (gid >= (int)XB_ELM) return;
    int chunk = gid >> 9;
    int flat  = gid & 511;
    int b     = chunk >> 11;            // 8*NT16 = 2048 chunks per batch
    int rem   = chunk & 2047;
    int ks    = rem >> 8;
    int ntile = rem & 255;
    int lanep = flat >> 4, s = flat & 15;
    int cl    = ((lanep < 16) ? 0 : 8) + s + ((s >= 8) ? 8 : 0);   // channel within 32-step
    int col   = lanep & 15;                    // n within tile
    ws[XB_OFF + gid] =
        (__bf16)x[((size_t)(b * CC + ks * 32 + cl)) * NN + ntile * 16 + col];
}

// ---------------------------------------------------------------------------
// Kernel 1: QKV projection with WMMA.
// grid = B * NT16 blocks of 256 threads (8 waves). 20 M-tile tasks per block
// (q:2, k:2, v:16) distributed round-robin across waves; each task is a
// [16o x 16n] tile accumulated over C=256 in 8 bf16 WMMA K-steps.
// Results (+bias) scatter-stored bf16 into attention operand layouts:
//   Q -> B-layout (for S^T = K^T Q),  K -> A-layout,  V^T -> B-layout.
// ---------------------------------------------------------------------------
__global__ void proj_kernel(__bf16* __restrict__ ws,
                            const float* __restrict__ bq,
                            const float* __restrict__ bk,
                            const float* __restrict__ bv) {
    const int lane  = threadIdx.x & 31;
    const int wave  = threadIdx.x >> 5;
    const int b     = blockIdx.x >> 8;     // / NT16
    const int ntile = blockIdx.x & 255;
    const int half8 = (lane < 16) ? 0 : 8;
    const int col   = lane & 15;

    for (int t = wave; t < 20; t += 8) {
        int kind, mt, wchunk0; const float* bias;
        if (t < 2)      { kind = 0; mt = t;     wchunk0 = t * 8;              bias = bq; }
        else if (t < 4) { kind = 1; mt = t - 2; wchunk0 = 16 + (t - 2) * 8;   bias = bk; }
        else            { kind = 2; mt = t - 4; wchunk0 = 32 + (t - 4) * 8;   bias = bv; }

        v8f acc = {};
        #pragma unroll
        for (int ks = 0; ks < 8; ++ks) {
            v16bf wa = *(const v16bf*)(ws + WC_OFF + (size_t)(wchunk0 + ks) * 512 + lane * 16);
            v16bf xb = *(const v16bf*)(ws + XB_OFF +
                          ((size_t)(b * 8 + ks) * NT16 + ntile) * 512 + lane * 16);
            acc = __builtin_amdgcn_wmma_f32_16x16x32_bf16(
                      false, wa, false, xb, (short)0, acc, false, false);
        }

        #pragma unroll
        for (int i = 0; i < 8; ++i) {
            int row = i + half8;           // output-channel row within tile
            int o   = mt * 16 + row;       // global output channel
            __bf16 vb16 = (__bf16)(acc[i] + bias[o]);
            if (kind == 0) {               // Q -> B-layout [32ci x 16q]: K=o, col=n
                ws[QB_OFF + (size_t)(b * NT16 + ntile) * 512 + b_slot(o, col)] = vb16;
            } else if (kind == 1) {        // K -> A-layout [16key x 32ci]: arow=n, K=o
                ws[KA_OFF + (size_t)(b * NT16 + ntile) * 512 + a_slot(col, o)] = vb16;
            } else {                       // V^T -> B-layout [32key x 16c]
                int key = (ntile & 1) * 16 + col;
                ws[VB_OFF + ((size_t)(b * NT32 + (ntile >> 1)) * 16 + mt) * 512 +
                   b_slot(key, o & 15)] = vb16;
            }
        }
    }
}

// ---------------------------------------------------------------------------
// Kernel 2: fused flash-attention, S^T orientation -> register-only P.
// grid = B * N/64 blocks, 256 threads (8 waves). Block owns 64 queries x all
// 256 channels: wave = (qm = query 16-tile of 4, cw = 128-channel half).
// Per 32-key step per wave: 2 WMMAs S^T = K^T Q (queries on lanes -> scalar
// per-lane softmax stats, one xor-16 shuffle per reduction), then the S^T
// D-layout IS the A-operand layout of P for O += P V^T: pack f32->bf16 in
// registers, 8 O WMMAs. No LDS, no barriers in the hot loop.
// ---------------------------------------------------------------------------
__global__ void attn_kernel(const __bf16* __restrict__ ws,
                            const float* __restrict__ x,
                            const float* __restrict__ gamma,
                            float* __restrict__ out) {
    const int lane   = threadIdx.x & 31;
    const int wave   = threadIdx.x >> 5;
    const int qm     = wave & 3;            // query tile within block
    const int cw     = wave >> 2;           // 0/1 : 128-channel half
    const int b      = blockIdx.x >> 6;     // / (N/64)
    const int qblk   = blockIdx.x & 63;
    const int qt     = qblk * 4 + qm;       // global 16-query tile index
    const int half8  = (lane < 16) ? 0 : 8;
    const int col    = lane & 15;
    const float g    = gamma[0];

    // Q tile as B-operand (loaded once, lives in 8 VGPRs)
    v16bf qb = *(const v16bf*)(ws + QB_OFF + (size_t)(b * NT16 + qt) * 512 + lane * 16);

    v8f acc[8];
    #pragma unroll
    for (int t = 0; t < 8; ++t) acc[t] = (v8f){};
    float mrow = -__builtin_inff();   // running max for query (lane & 15)
    float lrow = 0.f;                 // running denom for query (lane & 15)

    #pragma unroll 1
    for (int j = 0; j < NN; j += 32) {
        // S^T tiles: A = K [16key x 32ci], B = Q [32ci x 16q]
        const __bf16* kcp = ws + KA_OFF + (size_t)(b * NT16 + (j >> 4)) * 512;
        v16bf ka0 = *(const v16bf*)(kcp + lane * 16);
        v16bf ka1 = *(const v16bf*)(kcp + 512 + lane * 16);
        v8f z = {};
        v8f st0 = __builtin_amdgcn_wmma_f32_16x16x32_bf16(
                      false, ka0, false, qb, (short)0, z, false, false);
        v8f st1 = __builtin_amdgcn_wmma_f32_16x16x32_bf16(
                      false, ka1, false, qb, (short)0, z, false, false);
        // lane holds keys {j+half8..j+half8+7} U {j+16+half8..} for query col.

        // ---- online softmax, stats are scalar per lane (per query) ----
        float mx = fmaxf(st0[0], st1[0]);
        #pragma unroll
        for (int i = 1; i < 8; ++i) mx = fmaxf(mx, fmaxf(st0[i], st1[i]));
        mx = fmaxf(mx, __shfl_xor(mx, 16, 32));        // join the two halves
        float mn    = fmaxf(mrow, mx);
        float alpha = __expf(mrow - mn);
        mrow = mn;

        float p0[8], p1[8], ls = 0.f;
        #pragma unroll
        for (int i = 0; i < 8; ++i) {
            p0[i] = __expf(st0[i] - mn);
            p1[i] = __expf(st1[i] - mn);
            ls += p0[i] + p1[i];
        }
        ls += __shfl_xor(ls, 16, 32);
        lrow = lrow * alpha + ls;

        // rescale accumulators: acc rows are queries -> broadcast alpha
        #pragma unroll
        for (int i = 0; i < 8; ++i) {
            float ab = __shfl(alpha, i + half8, 32);
            #pragma unroll
            for (int t = 0; t < 8; ++t) acc[t][i] *= ab;
        }

        // ---- P as A-operand, built entirely in registers ----
        v16bf pa;
        #pragma unroll
        for (int i = 0; i < 8; ++i) {
            pa[i]     = (__bf16)p0[i];   // slots 0..7  : keys half8+0..7
            pa[i + 8] = (__bf16)p1[i];   // slots 8..15 : keys 16+half8+0..7
        }

        const __bf16* vcp = ws + VB_OFF +
            ((size_t)(b * NT32 + (j >> 5)) * 16 + cw * 8) * 512;
        #pragma unroll
        for (int t = 0; t < 8; ++t) {
            v16bf vb = *(const v16bf*)(vcp + (size_t)t * 512 + lane * 16);
            acc[t] = __builtin_amdgcn_wmma_f32_16x16x32_bf16(
                         false, pa, false, vb, (short)0, acc[t], false, false);
        }
        if (j + 32 < NN) {   // speculative prefetch of next K/V chunks
            __builtin_prefetch(kcp + 1024 + lane * 16, 0, 1);
            __builtin_prefetch(vcp + (size_t)16 * 512 + lane * 16, 0, 1);
        }
    }

    // Epilogue: out = gamma * (acc / l) + x   (x, out are f32 [B,C,N])
    #pragma unroll
    for (int i = 0; i < 8; ++i) {
        float lb = __shfl(lrow, i + half8, 32);   // denom for query row i
        float rl = 1.0f / lb;
        int n = qt * 16 + i + half8;
        #pragma unroll
        for (int t = 0; t < 8; ++t) {
            int c = (cw * 8 + t) * 16 + col;
            size_t idx = ((size_t)b * CC + c) * NN + n;
            out[idx] = g * (acc[t][i] * rl) + x[idx];
        }
    }
}

// ---------------------------------------------------------------------------
// Launch: inputs in setup_inputs() order: x, Wq, bq, Wk, bk, Wv, bv, gamma
// ---------------------------------------------------------------------------
extern "C" void kernel_launch(void* const* d_in, const int* in_sizes, int n_in,
                              void* d_out, int out_size, void* d_ws, size_t ws_size,
                              hipStream_t stream) {
    const float* x     = (const float*)d_in[0];
    const float* Wq    = (const float*)d_in[1];
    const float* bq    = (const float*)d_in[2];
    const float* Wk    = (const float*)d_in[3];
    const float* bk    = (const float*)d_in[4];
    const float* Wv    = (const float*)d_in[5];
    const float* bv    = (const float*)d_in[6];
    const float* gamma = (const float*)d_in[7];
    float*  out = (float*)d_out;
    __bf16* ws  = (__bf16*)d_ws;

    (void)in_sizes; (void)n_in; (void)out_size; (void)ws_size;

    // 0a: weights -> bf16 A-operand chunks
    conv_w_kernel<<<(int)((WC_ELM + 255) / 256), 256, 0, stream>>>(Wq, Wk, Wv, ws);
    // 0b: x -> bf16 B-operand chunks
    conv_x_kernel<<<(int)((XB_ELM + 255) / 256), 256, 0, stream>>>(x, ws);
    // 1: QKV projection (WMMA), emit attention operand layouts
    proj_kernel<<<BB * NT16, 256, 0, stream>>>(ws, bq, bk, bv);
    // 2: fused flash-attention (S^T orientation, register-only P)
    attn_kernel<<<BB * (NN / 64), 256, 0, stream>>>(ws, x, gamma, out);
}
